// NOGLSTM_16166256902983
// MI455X (gfx1250) — matrix operand order repaired and checked by reading, not compiled, not run on hardware
//
#include <hip/hip_runtime.h>
#include <stdint.h>
#include <stddef.h>

// ---------------- problem constants ----------------
#define BB   64          // batch
#define TT   512         // time steps
#define DD   256         // input dim
#define UU   512         // hidden units
#define G3   1536        // 3*UU (i|f|c gate columns concatenated)
#define RNB  32          // persistent workgroups in recurrent kernel (one u-tile each)

// ---------------- WMMA types (CDNA5, wave32) ----------------
typedef __attribute__((ext_vector_type(16))) __bf16   v16bf;
typedef __attribute__((ext_vector_type(8)))  float    v8f;
typedef __attribute__((ext_vector_type(4)))  unsigned u32x4;

union Frag {              // 16x32 (A) or 32x16 (B) bf16 fragment: 32B per lane
  v16bf v;
  u32x4 q[2];
};

__device__ __forceinline__ v8f wmma_bf16(const Frag& A, const Frag& B, v8f C) {
  // D = A(16x32 bf16) * B(32x16 bf16) + C(16x16 f32)
  return __builtin_amdgcn_wmma_f32_16x16x32_bf16(
      /*neg_a=*/false, A.v, /*neg_b=*/false, B.v,
      /*c_mod=*/(short)0, C, /*reuse_a=*/false, /*reuse_b=*/false);
}

// fp32 -> bf16 round-to-nearest-even (bit-level; avoids relying on __bf16 arith)
__device__ __forceinline__ unsigned short f2bf(float f) {
  union { float f; unsigned u; } cv; cv.f = f;
  unsigned u = cv.u;
  unsigned r = 0x7FFFu + ((u >> 16) & 1u);
  return (unsigned short)((u + r) >> 16);
}

__device__ __forceinline__ float sigmoidf_(float x) {
  return 1.0f / (1.0f + __expf(-x));
}
__device__ __forceinline__ float tanhf_(float x) {
  x = fminf(fmaxf(x, -15.0f), 15.0f);
  float e = __expf(2.0f * x);
  return (e - 1.0f) / (e + 1.0f);
}

// ---------------- grid-wide barrier (sense counter + generation) ----------------
__device__ __forceinline__ void grid_barrier(unsigned* cnt, unsigned* gen, unsigned nb) {
  __threadfence();
  __syncthreads();
  if (threadIdx.x == 0) {
    unsigned g = __hip_atomic_load(gen, __ATOMIC_RELAXED, __HIP_MEMORY_SCOPE_AGENT);
    unsigned a = __hip_atomic_fetch_add(cnt, 1u, __ATOMIC_ACQ_REL, __HIP_MEMORY_SCOPE_AGENT);
    if (a == nb - 1u) {
      __hip_atomic_store(cnt, 0u, __ATOMIC_RELAXED, __HIP_MEMORY_SCOPE_AGENT);
      __hip_atomic_fetch_add(gen, 1u, __ATOMIC_ACQ_REL, __HIP_MEMORY_SCOPE_AGENT);
    } else {
      while (__hip_atomic_load(gen, __ATOMIC_ACQUIRE, __HIP_MEMORY_SCOPE_AGENT) == g) {
        __builtin_amdgcn_s_sleep(2);
      }
    }
  }
  __syncthreads();
}

// ---------------- prep kernels ----------------
// x [B,T,D] f32 -> bf16 (row m = b*T+t, contiguous K=D)
__global__ void k_convert_x(const float* __restrict__ x, unsigned short* __restrict__ xbf) {
  int i = blockIdx.x * 256 + threadIdx.x;
  if (i < BB * TT * DD) xbf[i] = f2bf(x[i]);
}

// pack Wx (3x [D,U] f32) -> Wxt bf16 [G3, D], row j = g*U+n holds column n of gate g
__global__ void k_pack_wx(const float* __restrict__ wi, const float* __restrict__ wf,
                          const float* __restrict__ wc, unsigned short* __restrict__ Wxt) {
  int idx = blockIdx.x * 256 + threadIdx.x;          // (g*U+n)*D + k
  if (idx >= 3 * UU * DD) return;
  int g = idx / (UU * DD);
  int r = idx - g * (UU * DD);
  int n = r >> 8;            // D = 256
  int k = r & 255;
  const float* w = (g == 0) ? wi : ((g == 1) ? wf : wc);
  Wxt[idx] = f2bf(w[k * UU + n]);
}

// pack Wh (3x [U,U] f32) -> Wht bf16 [G3, U]
__global__ void k_pack_wh(const float* __restrict__ wi, const float* __restrict__ wf,
                          const float* __restrict__ wc, unsigned short* __restrict__ Wht) {
  int idx = blockIdx.x * 256 + threadIdx.x;          // (g*U+n)*U + k
  if (idx >= 3 * UU * UU) return;
  int g = idx >> 18;         // U*U = 262144
  int r = idx & 262143;
  int n = r >> 9;
  int k = r & 511;
  const float* w = (g == 0) ? wi : ((g == 1) ? wf : wc);
  Wht[idx] = f2bf(w[k * UU + n]);
}

// init state buffers from h0/c0; zero the grid barrier
__global__ void k_init(const float* __restrict__ h0, const float* __restrict__ c0,
                       unsigned short* __restrict__ hbuf0, float* __restrict__ cbuf,
                       unsigned* __restrict__ bar) {
  int i = blockIdx.x * 256 + threadIdx.x;
  if (i < BB * UU) {
    hbuf0[i] = f2bf(h0[i]);
    cbuf[i]  = c0[i];
  }
  if (blockIdx.x == 0 && threadIdx.x == 0) { bar[0] = 0u; bar[1] = 0u; }
}

// ---------------- kernel 1: batched input projection GEMM ----------------
// C[m, j] = sum_k xbf[m,k] * Wxt[j,k]  (m = b*T+t, j in [0,G3))
// stored into xg time-major: xg[(t*B + b)*G3 + j]
__global__ __launch_bounds__(128) void k_xproj(const unsigned short* __restrict__ xbf,
                                               const unsigned short* __restrict__ Wxt,
                                               float* __restrict__ xg) {
  const int wid  = threadIdx.x >> 5;
  const int lane = threadIdx.x & 31;
  const int hi   = lane >> 4;          // half-wave select
  const int lo   = lane & 15;
  const int tile = blockIdx.x * 4 + wid;       // 2048*96 tiles
  const int mt = tile / 96;
  const int nt = tile - mt * 96;

  const unsigned short* arow = xbf + (size_t)(mt * 16 + lo) * DD;
  const unsigned short* brow = Wxt + (size_t)(nt * 16 + lo) * DD;

  v8f c = {};
  for (int kt = 0; kt < DD / 32; ++kt) {
    const int k0 = kt * 32;
    Frag a, b;
    // A 16x32: lane-half hi gets 16 contiguous K starting at k0+16*hi
    const u32x4* ap = (const u32x4*)(arow + k0 + hi * 16);
    a.q[0] = ap[0];
    a.q[1] = ap[1];
    // B 32x16: lane-half hi gets 16 contiguous K starting at k0+16*hi
    const u32x4* bp = (const u32x4*)(brow + k0 + hi * 16);
    b.q[0] = bp[0];
    b.q[1] = bp[1];
    c = wmma_bf16(a, b, c);
  }

  // C layout: VGPR r -> row m0 + r + 8*hi, col n0 + lo
  const int col = nt * 16 + lo;
  for (int r = 0; r < 8; ++r) {
    int m = mt * 16 + r + hi * 8;
    int t = m & (TT - 1);
    int b = m >> 9;                    // T = 512
    xg[((size_t)t * BB + b) * G3 + col] = c[r];
  }
}

// ---------------- kernel 2: persistent fused recurrent scan ----------------
// 32 WGs x 128 thr (4 waves). WG = one 16-wide u-tile; wave = one 16-row batch tile.
// Per step: each wave computes i/f/c pre-activations for its 16x16 tile
// (3 gates x 16 K-tiles of v_wmma_f32_16x16x32_bf16, B from LDS, A = h from L2),
// then fuses the LSTM cell update in registers. One grid barrier per step;
// h ping-pongs between two bf16 buffers. Next step's xg segments are prefetched
// at the top of each step so HBM latency overlaps the WMMA chain + barrier.
__global__ __launch_bounds__(128) void k_recur(const unsigned short* __restrict__ Wht,
                                               const float* __restrict__ xg,
                                               unsigned short* __restrict__ hbuf0,
                                               unsigned short* __restrict__ hbuf1,
                                               float* __restrict__ cbuf,
                                               const float* __restrict__ bi,
                                               const float* __restrict__ bf,
                                               const float* __restrict__ bc,
                                               float* __restrict__ out,
                                               unsigned* __restrict__ bar) {
  __shared__ u32x4 bsh[3 * 16 * 64];   // 3 gates * 16 cols * 512 K bf16 = 48 KB

  const int u0  = blockIdx.x * 16;     // this WG's u-tile
  const int tid = threadIdx.x;

  // Stage this WG's weight slab into LDS once (rows g*U + u0 .. +15 are contiguous).
  for (int g = 0; g < 3; ++g) {
    const u32x4* src = (const u32x4*)(Wht + (size_t)(g * UU + u0) * UU);
    for (int i = tid; i < 1024; i += 128) bsh[g * 1024 + i] = src[i];
  }
  __syncthreads();

  const int wid  = tid >> 5;
  const int lane = tid & 31;
  const int hi   = lane >> 4;
  const int lo   = lane & 15;
  const int m0   = wid * 16;           // batch tile (B=64 -> 4 waves)
  const int u    = u0 + lo;

  const float bI = bi[u], bF = bf[u], bC = bc[u];
  const int rowq = lo * 64;            // lane's B column base, in u32x4 units

  // lane -> (row, gate) mapping for xg prefetch: 16 rows x 3 gates = 48 64B
  // segments per wave per step, covered by 2 prefetch instructions (lanes 0..23).
  const int pfr = lane / 3;            // 0..10 (only lanes 0..23 used)
  const int pfg = lane - pfr * 3;      // 0..2

  for (int t = 0; t < TT; ++t) {
    const unsigned short* hsrc = (t & 1) ? hbuf1 : hbuf0;
    unsigned short*       hdst = (t & 1) ? hbuf0 : hbuf1;

    // Prefetch next step's gate-activation segments (read-only stream from HBM).
    if (t + 1 < TT && lane < 24) {
      const float* base = xg + (size_t)(t + 1) * BB * G3 + (size_t)pfg * UU + u0;
      __builtin_prefetch(base + (size_t)(m0 + pfr) * G3, 0, 3);
      __builtin_prefetch(base + (size_t)(m0 + 8 + pfr) * G3, 0, 3);
    }

    v8f ci = {}, cf = {}, cc = {};
    const unsigned short* arow = hsrc + (size_t)(m0 + lo) * UU;

    for (int kt = 0; kt < UU / 32; ++kt) {
      const int k0 = kt * 32;
      Frag a;
      const u32x4* ap = (const u32x4*)(arow + k0 + hi * 16);
      a.q[0] = ap[0];
      a.q[1] = ap[1];

      const int qb = (k0 + hi * 16) >> 3;     // u32x4 offset within a 512-K row
      Frag b0, b1, b2;
      b0.q[0] = bsh[rowq + qb];          b0.q[1] = bsh[rowq + qb + 1];
      b1.q[0] = bsh[1024 + rowq + qb];   b1.q[1] = bsh[1024 + rowq + qb + 1];
      b2.q[0] = bsh[2048 + rowq + qb];   b2.q[1] = bsh[2048 + rowq + qb + 1];

      ci = wmma_bf16(a, b0, ci);
      cf = wmma_bf16(a, b1, cf);
      cc = wmma_bf16(a, b2, cc);
    }

    // Fused LSTM cell update — accumulators never leave registers.
    for (int r = 0; r < 8; ++r) {
      const int b = m0 + r + hi * 8;                        // batch row
      const float* xr = xg + ((size_t)t * BB + b) * G3;
      float gi  = sigmoidf_(xr[u]           + bI + ci[r]);
      float gf  = sigmoidf_(xr[UU + u]      + bF + cf[r]);
      float gc  = tanhf_   (xr[2 * UU + u]  + bC + cc[r]);
      const int e = b * UU + u;
      float cn = gf * cbuf[e] + gi * gc;
      float hn = tanhf_(cn);
      cbuf[e] = cn;
      hdst[e] = f2bf(hn);
      // streaming output: non-temporal so it doesn't evict xg from L2
      __builtin_nontemporal_store(hn, out + (size_t)b * (TT * UU) + (size_t)t * UU + u);
    }

    grid_barrier(bar, bar + 1, RNB);     // h(t+1) visible before next step reads it
  }
}

// ---------------- host launcher ----------------
extern "C" void kernel_launch(void* const* d_in, const int* in_sizes, int n_in,
                              void* d_out, int out_size, void* d_ws, size_t ws_size,
                              hipStream_t stream) {
  (void)in_sizes; (void)n_in; (void)out_size; (void)ws_size;

  const float* x   = (const float*)d_in[0];
  const float* wxi = (const float*)d_in[1];
  const float* wxf = (const float*)d_in[2];
  const float* wxc = (const float*)d_in[3];
  const float* whi = (const float*)d_in[4];
  const float* whf = (const float*)d_in[5];
  const float* whc = (const float*)d_in[6];
  const float* bi  = (const float*)d_in[7];
  const float* bf  = (const float*)d_in[8];
  const float* bc  = (const float*)d_in[9];
  const float* h0  = (const float*)d_in[10];
  const float* c0  = (const float*)d_in[11];
  float* out = (float*)d_out;

  // workspace layout
  char* w = (char*)d_ws;
  size_t off = 0;
  unsigned* bar = (unsigned*)(w + off);                 off += 256;
  unsigned short* xbf = (unsigned short*)(w + off);     off += (size_t)BB * TT * DD * 2;   // 16 MB
  unsigned short* Wxt = (unsigned short*)(w + off);     off += (size_t)G3 * DD * 2;        // 768 KB
  unsigned short* Wht = (unsigned short*)(w + off);     off += (size_t)G3 * UU * 2;        // 1.5 MB
  unsigned short* hb0 = (unsigned short*)(w + off);     off += (size_t)BB * UU * 2;
  unsigned short* hb1 = (unsigned short*)(w + off);     off += (size_t)BB * UU * 2;
  float* cbuf = (float*)(w + off);                      off += (size_t)BB * UU * 4;
  float* xg   = (float*)(w + off);                      off += (size_t)TT * BB * G3 * 4;   // 192 MB

  k_init<<<(BB * UU + 255) / 256, 256, 0, stream>>>(h0, c0, hb0, cbuf, bar);
  k_convert_x<<<(BB * TT * DD + 255) / 256, 256, 0, stream>>>(x, xbf);
  k_pack_wx<<<(3 * UU * DD + 255) / 256, 256, 0, stream>>>(wxi, wxf, wxc, Wxt);
  k_pack_wh<<<(3 * UU * UU + 255) / 256, 256, 0, stream>>>(whi, whf, whc, Wht);

  // 32768/16 = 2048 m-tiles x 96 n-tiles, 4 waves (tiles) per 128-thread WG
  k_xproj<<<(2048 * 96) / 4, 128, 0, stream>>>(xbf, Wxt, xg);

  // persistent fused recurrent scan: 32 WGs x 128 threads
  k_recur<<<RNB, 128, 0, stream>>>(Wht, xg, hb0, hb1, cbuf, bi, bf, bc, out, bar);
}
